// Network_43920335569257
// MI455X (gfx1250) — matrix operand order repaired and checked by reading, not compiled, not run on hardware
//
#include <hip/hip_runtime.h>

typedef __attribute__((ext_vector_type(16))) _Float16 v16h;
typedef __attribute__((ext_vector_type(8)))  _Float16 v8h;
typedef __attribute__((ext_vector_type(4)))  _Float16 v4h;
typedef __attribute__((ext_vector_type(8)))  float    v8f;
typedef __attribute__((ext_vector_type(4)))  float    v4f;

#define LDT 40   // LDS tile row stride in halves (80 B: multiple of 16 B, conflict-free)

enum { EPI_RELU_F16 = 0, EPI_QUANT_F16 = 1, EPI_NONE_F32 = 2 };

// ---------------------------------------------------------------------------
// fp32 -> fp16 flat conversion (vectorized x4) for activations
// ---------------------------------------------------------------------------
__global__ __launch_bounds__(256) void cvt_f32_to_f16(const float* __restrict__ in,
                                                      _Float16* __restrict__ out,
                                                      long long n4) {
  long long i = (long long)blockIdx.x * blockDim.x + threadIdx.x;
  if (i >= n4) return;
  v4f v = ((const v4f*)in)[i];
  v4h h;
  for (int j = 0; j < 4; ++j) h[j] = (_Float16)v[j];
  ((v4h*)out)[i] = h;
}

// ---------------------------------------------------------------------------
// fp32 [K,N] -> fp16 [N,K] transposed conversion for weights (one-time prep,
// so the GEMM can stage B tiles with contiguous 128-bit LDS stores).
// ---------------------------------------------------------------------------
__global__ __launch_bounds__(256) void transpose_cvt_f32_to_f16(
    const float* __restrict__ in, _Float16* __restrict__ out, int K, int N) {
  __shared__ float t[32][33];
  const int tx = threadIdx.x & 31;
  const int ty = threadIdx.x >> 5;  // 8 row-groups
  const int k0 = blockIdx.y * 32;
  const int n0 = blockIdx.x * 32;
  #pragma unroll
  for (int r = ty; r < 32; r += 8) t[r][tx] = in[(size_t)(k0 + r) * N + (n0 + tx)];
  __syncthreads();
  #pragma unroll
  for (int r = ty; r < 32; r += 8)
    out[(size_t)(n0 + r) * K + (k0 + tx)] = (_Float16)t[tx][r];
}

// ---------------------------------------------------------------------------
// Tiled WMMA GEMM: C[M,N] = A[M,K] (f16, row-major) x Bt[N,K]^T (f16) + bias.
// 256 threads = 8 wave32s; block tile 128x128, wave tile 32x64, K chunk 32.
// Double-buffered LDS (one barrier per K-chunk), f32 accumulation.
// ---------------------------------------------------------------------------
template <int EPI>
__global__ __launch_bounds__(256) void gemm_wmma_f16(
    const _Float16* __restrict__ A,   // [M,K] row-major f16
    const _Float16* __restrict__ Bt,  // [N,K] row-major f16 (W transposed)
    const float* __restrict__ bias,   // [N] f32
    const float* __restrict__ cbook,  // [16] f32 (EPI_QUANT_F16 only)
    _Float16* __restrict__ Oh,        // f16 output (EPI_RELU/QUANT)
    float* __restrict__ Of,           // f32 output (EPI_NONE_F32)
    int M, int N, int K) {
  __shared__ _Float16 As[2][128 * LDT];  // [m][k] tiles, ping-pong
  __shared__ _Float16 Bs[2][128 * LDT];  // [n][k] tiles, ping-pong

  const int tid  = threadIdx.x;
  const int lane = tid & 31;
  const int wave = tid >> 5;
  const int m0 = blockIdx.y * 128;
  const int n0 = blockIdx.x * 128;
  const int wm = (wave >> 1) * 32;  // wave M offset in tile: 0/32/64/96
  const int wn = (wave & 1) * 64;   // wave N offset in tile: 0/64
  const int l15 = lane & 15;
  const int kA  = (lane & 16) ? 8  : 0;  // A frag: halves {kA..kA+7, kA+16..kA+23}
  const int kB  = (lane & 16) ? 8  : 0;  // B frag: halves {kB..kB+7, kB+16..kB+23}

  float cb[16];
  if (EPI == EPI_QUANT_F16) {
    #pragma unroll
    for (int i = 0; i < 16; ++i) cb[i] = cbook[i];
  }

  // staging: each thread moves 16 halves of A and 16 halves of B per chunk
  const int srow = tid >> 1;        // 0..127 (m for A, n for B)
  const int scol = (tid & 1) * 16;  // 0 or 16 (k)
  const _Float16* aRow = A  + (size_t)(m0 + srow) * K + scol;
  const _Float16* bRow = Bt + (size_t)(n0 + srow) * K + scol;
  const int sOff = srow * LDT + scol;

  v8f acc[2][4] = {};

  // preload chunk 0 into buffer 0
  v8h a0 = *(const v8h*)(aRow);
  v8h a1 = *(const v8h*)(aRow + 8);
  v8h b0 = *(const v8h*)(bRow);
  v8h b1 = *(const v8h*)(bRow + 8);
  *(v8h*)(&As[0][sOff])     = a0;
  *(v8h*)(&As[0][sOff + 8]) = a1;
  *(v8h*)(&Bs[0][sOff])     = b0;
  *(v8h*)(&Bs[0][sOff + 8]) = b1;
  __syncthreads();

  int p = 0;
  for (int k0 = 0; k0 < K; k0 += 32) {
    const bool more = (k0 + 32) < K;
    if (more) {  // fetch chunk k0+32 while computing chunk k0
      const _Float16* ag = aRow + k0 + 32;
      const _Float16* bg = bRow + k0 + 32;
      a0 = *(const v8h*)(ag);
      a1 = *(const v8h*)(ag + 8);
      b0 = *(const v8h*)(bg);
      b1 = *(const v8h*)(bg + 8);
      if (k0 + 64 < K) {  // prefetch chunk k0+64 -> global_prefetch_b8
        __builtin_prefetch((const void*)(ag + 32), 0, 1);
        __builtin_prefetch((const void*)(bg + 32), 0, 1);
      }
    }

    // fragment loads (16 B ds_load_b128 each), per documented CDNA5 layouts
    v16h af[2], bf[4];
    #pragma unroll
    for (int mi = 0; mi < 2; ++mi) {
      const _Float16* q = &As[p][(wm + mi * 16 + l15) * LDT + kA];
      v8h lo = *(const v8h*)(q);
      v8h hi = *(const v8h*)(q + 16);
      af[mi] = __builtin_shufflevector(lo, hi, 0, 1, 2, 3, 4, 5, 6, 7,
                                               8, 9, 10, 11, 12, 13, 14, 15);
    }
    #pragma unroll
    for (int ni = 0; ni < 4; ++ni) {
      const _Float16* q = &Bs[p][(wn + ni * 16 + l15) * LDT + kB];
      v8h lo = *(const v8h*)(q);
      v8h hi = *(const v8h*)(q + 16);
      bf[ni] = __builtin_shufflevector(lo, hi, 0, 1, 2, 3, 4, 5, 6, 7,
                                               8, 9, 10, 11, 12, 13, 14, 15);
    }

    #pragma unroll
    for (int mi = 0; mi < 2; ++mi)
      #pragma unroll
      for (int ni = 0; ni < 4; ++ni)
        acc[mi][ni] = __builtin_amdgcn_wmma_f32_16x16x32_f16(
            false, af[mi], false, bf[ni], (short)0, acc[mi][ni], false, false);

    if (more) {  // stage chunk k0+32 into the other buffer
      *(v8h*)(&As[p ^ 1][sOff])     = a0;
      *(v8h*)(&As[p ^ 1][sOff + 8]) = a1;
      *(v8h*)(&Bs[p ^ 1][sOff])     = b0;
      *(v8h*)(&Bs[p ^ 1][sOff + 8]) = b1;
    }
    __syncthreads();
    p ^= 1;
  }

  // epilogue: C/D layout -> lane L: N = nBase + (L&15), rows rBase..rBase+7
  const int rBase = (lane & 16) ? 8 : 0;
  #pragma unroll
  for (int ni = 0; ni < 4; ++ni) {
    const int col = n0 + wn + ni * 16 + l15;
    const float bv = bias[col];
    #pragma unroll
    for (int mi = 0; mi < 2; ++mi) {
      #pragma unroll
      for (int r = 0; r < 8; ++r) {
        const int row = m0 + wm + mi * 16 + rBase + r;
        float v = acc[mi][ni][r] + bv;
        if (EPI == EPI_RELU_F16) v = v > 0.0f ? v : 0.0f;
        if (EPI == EPI_QUANT_F16) {
          float best = cb[0];
          float bd = fabsf(v - cb[0]);
          #pragma unroll
          for (int i = 1; i < 16; ++i) {  // strict < keeps lowest index (argmin)
            float d = fabsf(v - cb[i]);
            if (d < bd) { bd = d; best = cb[i]; }
          }
          v = best;
        }
        if (EPI == EPI_NONE_F32) Of[(size_t)row * N + col] = v;
        else                     Oh[(size_t)row * N + col] = (_Float16)v;
      }
    }
  }
}

// ---------------------------------------------------------------------------
// Launcher
// ---------------------------------------------------------------------------
extern "C" void kernel_launch(void* const* d_in, const int* in_sizes, int n_in,
                              void* d_out, int out_size, void* d_ws, size_t ws_size,
                              hipStream_t stream) {
  const float* x  = (const float*)d_in[0];
  const float* cb = (const float*)d_in[1];
  const float* W1 = (const float*)d_in[2];
  const float* b1 = (const float*)d_in[3];
  const float* W2 = (const float*)d_in[4];
  const float* b2 = (const float*)d_in[5];
  const float* W3 = (const float*)d_in[6];
  const float* b3 = (const float*)d_in[7];
  const float* W4 = (const float*)d_in[8];
  const float* b4 = (const float*)d_in[9];
  float* out = (float*)d_out;

  const int B = 32768, DIN = 1024, DOUT = 512;

  char* ws = (char*)d_ws;
  size_t off = 0;
  _Float16* xh  = (_Float16*)(ws + off); off += (size_t)B * DIN * 2;   // 64 MB
  _Float16* h1h = (_Float16*)(ws + off); off += (size_t)B * DIN * 2;   // 64 MB
  _Float16* h2h = (_Float16*)(ws + off); off += (size_t)B * DOUT * 2;  // 32 MB
  _Float16* W1t = (_Float16*)(ws + off); off += (size_t)DIN * DIN * 2;   // [N,K]=[1024,1024]
  _Float16* W2t = (_Float16*)(ws + off); off += (size_t)DIN * DOUT * 2;  // [512,1024]
  _Float16* W3t = (_Float16*)(ws + off); off += (size_t)DOUT * DOUT * 2; // [512,512]
  _Float16* W4t = (_Float16*)(ws + off); off += (size_t)DOUT * DOUT * 2; // [512,512]
  _Float16* h3h = xh;  // x no longer needed after layer 1

  // prep: x -> f16 flat; W -> f16 transposed [N,K]
  {
    long long n4 = (long long)B * DIN / 4;
    cvt_f32_to_f16<<<dim3((unsigned)((n4 + 255) / 256)), dim3(256), 0, stream>>>(x, xh, n4);
  }
  transpose_cvt_f32_to_f16<<<dim3(DIN / 32, DIN / 32),  dim3(256), 0, stream>>>(W1, W1t, DIN, DIN);
  transpose_cvt_f32_to_f16<<<dim3(DOUT / 32, DIN / 32), dim3(256), 0, stream>>>(W2, W2t, DIN, DOUT);
  transpose_cvt_f32_to_f16<<<dim3(DOUT / 32, DOUT / 32), dim3(256), 0, stream>>>(W3, W3t, DOUT, DOUT);
  transpose_cvt_f32_to_f16<<<dim3(DOUT / 32, DOUT / 32), dim3(256), 0, stream>>>(W4, W4t, DOUT, DOUT);

  dim3 blk(256);
  // h1 = relu(x @ W1 + b1)           [32768,1024]
  gemm_wmma_f16<EPI_RELU_F16><<<dim3(DIN / 128, B / 128), blk, 0, stream>>>(
      xh, W1t, b1, nullptr, h1h, nullptr, B, DIN, DIN);
  // h2 = quantize(h1 @ W2 + b2)      [32768,512]
  gemm_wmma_f16<EPI_QUANT_F16><<<dim3(DOUT / 128, B / 128), blk, 0, stream>>>(
      h1h, W2t, b2, cb, h2h, nullptr, B, DOUT, DIN);
  // h3 = relu(h2 @ W3 + b3)          [32768,512]
  gemm_wmma_f16<EPI_RELU_F16><<<dim3(DOUT / 128, B / 128), blk, 0, stream>>>(
      h2h, W3t, b3, nullptr, h3h, nullptr, B, DOUT, DOUT);
  // out = h3 @ W4 + b4               [32768,512] f32
  gemm_wmma_f16<EPI_NONE_F32><<<dim3(DOUT / 128, B / 128), blk, 0, stream>>>(
      h3h, W4t, b4, nullptr, nullptr, out, B, DOUT, DOUT);
}
